// GATModelExtended_20993800143363
// MI455X (gfx1250) — matrix operand order repaired and checked by reading, not compiled, not run on hardware
//
#include <hip/hip_runtime.h>

// ---------------- problem constants (match reference) ----------------
#define NN      8192            // nodes
#define EE      65536           // edges (no self loops)
#define ETOT    (EE + NN)       // edges incl. self loops
#define GG      64              // graphs
#define DNODE   64
#define DEDGE   32
#define CCH     256
#define NH1     8
#define NH2     4
#define FF1     (NH1 * CCH)     // 2048
#define FF2     (NH2 * CCH)     // 1024
#define NEG_SLOPE 0.2f

typedef __attribute__((ext_vector_type(16))) __bf16 v16bf;
typedef __attribute__((ext_vector_type(8)))  __bf16 v8bf;
typedef __attribute__((ext_vector_type(8)))  float  v8f;

// ---------------- helpers ----------------
__device__ __forceinline__ unsigned short f2bf(float f) {
  unsigned int u = __float_as_uint(f);
  u += 0x7FFFu + ((u >> 16) & 1u);          // round-to-nearest-even
  return (unsigned short)(u >> 16);
}
// monotone float <-> uint key so unsigned atomicMax == float max
__device__ __forceinline__ unsigned int fkey(float f) {
  unsigned int b = __float_as_uint(f);
  return b ^ ((b & 0x80000000u) ? 0xFFFFFFFFu : 0x80000000u);
}
__device__ __forceinline__ float funkey(unsigned int k) {
  unsigned int b = (k & 0x80000000u) ? (k ^ 0x80000000u) : ~k;
  return __uint_as_float(b);
}

// ---------------- generic utility kernels ----------------
__global__ void zero_f32(float* __restrict__ p, size_t n) {
  size_t i = (size_t)blockIdx.x * blockDim.x + threadIdx.x;
  if (i < n) p[i] = 0.0f;
}

__global__ void f32_to_bf16(const float* __restrict__ in, unsigned short* __restrict__ out, size_t n) {
  size_t i = (size_t)blockIdx.x * blockDim.x + threadIdx.x;
  if (i < n) out[i] = f2bf(in[i]);
}

// ---------------- self-loop attr = mean of incoming edge attrs ----------------
__global__ void selfloop_accum(const int* __restrict__ dstI, const float* __restrict__ ea,
                               float* __restrict__ loop_ea, float* __restrict__ deg) {
  int idx = blockIdx.x * blockDim.x + threadIdx.x;        // E*32 threads
  if (idx >= EE * DEDGE) return;
  int e = idx >> 5, c = idx & 31;
  int d = dstI[e];
  atomicAdd(loop_ea + (size_t)d * DEDGE + c, ea[(size_t)e * DEDGE + c]);
  if (c == 0) atomicAdd(deg + d, 1.0f);
}

__global__ void selfloop_div(float* __restrict__ loop_ea, const float* __restrict__ deg) {
  int idx = blockIdx.x * blockDim.x + threadIdx.x;        // N*32 threads
  if (idx >= NN * DEDGE) return;
  loop_ea[idx] /= fmaxf(deg[idx >> 5], 1.0f);
}

// ---------------- bf16 WMMA GEMM:  D[M,Nout] = A[M,K] @ W[Nout,K]^T + bias ----------------
// 16x64 register tile per wave: one A fragment feeds 4 B fragments / 4 accumulators.
// Software-pipelined: next K-slab fragments are fetched before the current 4 WMMAs issue.
__global__ void gemm_bf16_wmma(const unsigned short* __restrict__ A,
                               const unsigned short* __restrict__ W,
                               const float* __restrict__ bias,
                               float* __restrict__ D,
                               int M, int Nout, int K, int relu) {
  int wave   = blockIdx.x * (blockDim.x >> 5) + (threadIdx.x >> 5);
  int ntn    = Nout >> 6;                       // 64-wide column tiles
  int ntiles = (M >> 4) * ntn;
  if (wave >= ntiles) return;                   // whole-wave uniform: EXEC all-1s inside
  int tm = wave / ntn, tn = wave % ntn;
  int l = threadIdx.x & 31;
  int half = l >> 4, lr = l & 15;

  const unsigned short* arow = A + (size_t)(tm * 16 + lr) * K;       // A: 16-bit 16x32 layout
  const unsigned short* br0  = W + (size_t)(tn * 64 + lr) * K;       // B col n == W row n
  const unsigned short* br1  = br0 + (size_t)16 * K;
  const unsigned short* br2  = br0 + (size_t)32 * K;
  const unsigned short* br3  = br0 + (size_t)48 * K;

#define LOAD_A(k_) ({                                                              \
    v8bf lo_ = *(const v8bf*)(arow + (k_) + half * 8);                             \
    v8bf hi_ = *(const v8bf*)(arow + (k_) + 16 + half * 8);                        \
    __builtin_shufflevector(lo_, hi_, 0,1,2,3,4,5,6,7,8,9,10,11,12,13,14,15); })
#define LOAD_B(p_, k_) ({                                                          \
    v8bf blo_ = *(const v8bf*)((p_) + (k_) + half * 8);                            \
    v8bf bhi_ = *(const v8bf*)((p_) + (k_) + 16 + half * 8);                       \
    __builtin_shufflevector(blo_, bhi_, 0,1,2,3,4,5,6,7,8,9,10,11,12,13,14,15); })

  v8f acc0 = {}, acc1 = {}, acc2 = {}, acc3 = {};
  v16bf av  = LOAD_A(0);
  v16bf bv0 = LOAD_B(br0, 0), bv1 = LOAD_B(br1, 0);
  v16bf bv2 = LOAD_B(br2, 0), bv3 = LOAD_B(br3, 0);

  for (int k = 0; k < K; k += 32) {
    int kn = (k + 32 < K) ? (k + 32) : k;       // last iter: harmless redundant refetch
    if (k + 64 < K) {                           // long-range L2 warm-up (global_prefetch_b8)
      __builtin_prefetch((const void*)(arow + k + 64), 0, 1);
      __builtin_prefetch((const void*)(br0 + k + 64), 0, 1);
      __builtin_prefetch((const void*)(br1 + k + 64), 0, 1);
      __builtin_prefetch((const void*)(br2 + k + 64), 0, 1);
      __builtin_prefetch((const void*)(br3 + k + 64), 0, 1);
    }
    // fetch next slab first so the 4 WMMAs below overlap these loads
    v16bf av_n  = LOAD_A(kn);
    v16bf bv0_n = LOAD_B(br0, kn), bv1_n = LOAD_B(br1, kn);
    v16bf bv2_n = LOAD_B(br2, kn), bv3_n = LOAD_B(br3, kn);

    acc0 = __builtin_amdgcn_wmma_f32_16x16x32_bf16(false, av, false, bv0, (short)0, acc0, false, false);
    acc1 = __builtin_amdgcn_wmma_f32_16x16x32_bf16(false, av, false, bv1, (short)0, acc1, false, false);
    acc2 = __builtin_amdgcn_wmma_f32_16x16x32_bf16(false, av, false, bv2, (short)0, acc2, false, false);
    acc3 = __builtin_amdgcn_wmma_f32_16x16x32_bf16(false, av, false, bv3, (short)0, acc3, false, false);

    av = av_n; bv0 = bv0_n; bv1 = bv1_n; bv2 = bv2_n; bv3 = bv3_n;
  }
#undef LOAD_A
#undef LOAD_B

  v8f accs[4] = {acc0, acc1, acc2, acc3};
#pragma unroll
  for (int cb = 0; cb < 4; ++cb) {
    int col = tn * 64 + cb * 16 + lr;
    float bb = bias ? bias[col] : 0.0f;
#pragma unroll
    for (int v = 0; v < 8; ++v) {               // D: lane=col, vgpr v -> row v + 8*half
      int row = tm * 16 + v + half * 8;
      float o = accs[cb][v] + bb;
      if (relu) o = fmaxf(o, 0.0f);
      D[(size_t)row * Nout + col] = o;
    }
  }
}

// ---------------- GATv2 edge kernels ----------------
// One wave per (edge, head). ee = ea_row @ We_rows computed on the fly (fused, never stored).
__global__ void gat_logits(const float* __restrict__ xl, const float* __restrict__ xr,
                           const int* __restrict__ srcI, const int* __restrict__ dstI,
                           const float* __restrict__ ea, const float* __restrict__ loop_ea,
                           const float* __restrict__ We, const float* __restrict__ att,
                           float* __restrict__ logits, unsigned int* __restrict__ mxkey,
                           int H, int F) {
  int wid = (blockIdx.x * blockDim.x + threadIdx.x) >> 5;
  if (wid >= ETOT * H) return;
  int l = threadIdx.x & 31;
  int e = wid / H, h = wid % H;
  int s_n = (e < EE) ? srcI[e] : (e - EE);
  int d_n = (e < EE) ? dstI[e] : (e - EE);
  const float* erow = (e < EE) ? (ea + (size_t)e * DEDGE) : (loop_ea + (size_t)(e - EE) * DEDGE);
  float eav = erow[l];                                   // lane l holds edge-attr element l
  int cbase = h * CCH + l * 8;                           // 8 channels per lane
  const float* xlp = xl + (size_t)s_n * F + cbase;
  const float* xrp = xr + (size_t)d_n * F + cbase;
  float s[8];
#pragma unroll
  for (int j = 0; j < 8; ++j) s[j] = xlp[j] + xrp[j];
  const float* wb = We + (size_t)cbase * DEDGE;          // We[(h*C+c), 32]
  for (int k = 0; k < DEDGE; ++k) {
    float a = __shfl(eav, k, 32);                        // broadcast ea[e,k]
#pragma unroll
    for (int j = 0; j < 8; ++j) s[j] = fmaf(a, wb[j * DEDGE + k], s[j]);
  }
  float partial = 0.0f;
  const float* ap = att + cbase;
#pragma unroll
  for (int j = 0; j < 8; ++j) {
    float m = s[j] > 0.0f ? s[j] : s[j] * NEG_SLOPE;     // leaky relu
    partial = fmaf(m, ap[j], partial);
  }
#pragma unroll
  for (int o = 16; o > 0; o >>= 1) partial += __shfl_xor(partial, o, 32);
  if (l == 0) {
    logits[wid] = partial;
    atomicMax(mxkey + (size_t)d_n * H + h, fkey(partial));
  }
}

__global__ void gat_exps(float* __restrict__ logits, const unsigned int* __restrict__ mxkey,
                         float* __restrict__ den, const int* __restrict__ dstI, int H) {
  int wid = blockIdx.x * blockDim.x + threadIdx.x;
  if (wid >= ETOT * H) return;
  int e = wid / H, h = wid % H;
  int d_n = (e < EE) ? dstI[e] : (e - EE);
  float ex = __expf(logits[wid] - funkey(mxkey[(size_t)d_n * H + h]));
  logits[wid] = ex;
  atomicAdd(den + (size_t)d_n * H + h, ex);
}

__global__ void gat_accum(const float* __restrict__ xl, const float* __restrict__ ex,
                          const float* __restrict__ den, const int* __restrict__ srcI,
                          const int* __restrict__ dstI, float* __restrict__ out, int H, int F) {
  int wid = (blockIdx.x * blockDim.x + threadIdx.x) >> 5;
  if (wid >= ETOT * H) return;
  int l = threadIdx.x & 31;
  int e = wid / H, h = wid % H;
  int s_n = (e < EE) ? srcI[e] : (e - EE);
  int d_n = (e < EE) ? dstI[e] : (e - EE);
  float alpha = ex[wid] / den[(size_t)d_n * H + h];
  int cbase = h * CCH + l * 8;
  const float* xp = xl + (size_t)s_n * F + cbase;
  float* op = out + (size_t)d_n * F + cbase;
#pragma unroll
  for (int j = 0; j < 8; ++j) atomicAdd(op + j, alpha * xp[j]);   // global_atomic_add_f32 (L2)
}

__global__ void bias_relu_bf16(const float* __restrict__ in, const float* __restrict__ bias,
                               unsigned short* __restrict__ out, size_t n, int F) {
  size_t i = (size_t)blockIdx.x * blockDim.x + threadIdx.x;
  if (i >= n) return;
  float v = fmaxf(in[i] + bias[i % F], 0.0f);
  out[i] = f2bf(v);
}

__global__ void bias_add_inplace(float* __restrict__ a, const float* __restrict__ bias,
                                 size_t n, int F) {
  size_t i = (size_t)blockIdx.x * blockDim.x + threadIdx.x;
  if (i < n) a[i] += bias[i % F];
}

// ---------------- pooling + MLP ----------------
__global__ void pool_cnt(const int* __restrict__ batch, float* __restrict__ cnt) {
  int n = blockIdx.x * blockDim.x + threadIdx.x;
  if (n < NN) atomicAdd(cnt + batch[n], 1.0f);
}
__global__ void pool_sum(const float* __restrict__ h, const int* __restrict__ batch,
                         float* __restrict__ pooled) {
  size_t i = (size_t)blockIdx.x * blockDim.x + threadIdx.x;
  if (i >= (size_t)NN * FF2) return;
  int n = (int)(i / FF2), c = (int)(i % FF2);
  atomicAdd(pooled + (size_t)batch[n] * FF2 + c, h[i]);
}
__global__ void pool_div(float* __restrict__ pooled, const float* __restrict__ cnt) {
  int i = blockIdx.x * blockDim.x + threadIdx.x;
  if (i < GG * FF2) pooled[i] /= fmaxf(cnt[i / FF2], 1.0f);
}

__global__ void linear_act(const float* __restrict__ X, const float* __restrict__ W,
                           const float* __restrict__ b, float* __restrict__ Y,
                           int Gn, int K, int J, int act) {
  int idx = blockIdx.x * blockDim.x + threadIdx.x;
  if (idx >= Gn * J) return;
  int g = idx / J, j = idx % J;
  const float* x = X + (size_t)g * K;
  const float* w = W + (size_t)j * K;
  float s = b[j];
  for (int k = 0; k < K; ++k) s = fmaf(x[k], w[k], s);
  if (act == 1) s = fmaxf(s, 0.0f);
  else if (act == 2) s = 1.0f / (1.0f + __expf(-s));
  Y[(size_t)g * J + j] = s;
}

// ---------------- host orchestration ----------------
static inline unsigned cdiv(size_t n, int b) { return (unsigned)((n + b - 1) / b); }

extern "C" void kernel_launch(void* const* d_in, const int* in_sizes, int n_in,
                              void* d_out, int out_size, void* d_ws, size_t ws_size,
                              hipStream_t stream) {
  const float* x     = (const float*)d_in[0];
  const int*   ei    = (const int*)d_in[1];
  const float* ea    = (const float*)d_in[2];
  const int*   batch = (const int*)d_in[3];
  const float* Wl1 = (const float*)d_in[4];  const float* bl1 = (const float*)d_in[5];
  const float* Wr1 = (const float*)d_in[6];  const float* br1 = (const float*)d_in[7];
  const float* We1 = (const float*)d_in[8];  const float* att1 = (const float*)d_in[9];
  const float* bias1 = (const float*)d_in[10];
  const float* Wl2 = (const float*)d_in[11]; const float* bl2 = (const float*)d_in[12];
  const float* Wr2 = (const float*)d_in[13]; const float* br2 = (const float*)d_in[14];
  const float* We2 = (const float*)d_in[15]; const float* att2 = (const float*)d_in[16];
  const float* bias2 = (const float*)d_in[17];
  const float* W1 = (const float*)d_in[18];  const float* b1 = (const float*)d_in[19];
  const float* W2 = (const float*)d_in[20];  const float* b2 = (const float*)d_in[21];
  const float* W3 = (const float*)d_in[22];  const float* b3 = (const float*)d_in[23];
  const int* srcI = ei;
  const int* dstI = ei + EE;

  char* ws = (char*)d_ws;
  size_t off = 0;
  auto take = [&](size_t bytes) { size_t o = off; off += (bytes + 255) & ~(size_t)255; return o; };

  float* xl1   = (float*)(ws + take((size_t)NN * FF1 * 4));   // 64MB (reused: xl2, xr2)
  float* xr1   = (float*)(ws + take((size_t)NN * FF1 * 4));   // 64MB
  float* outb  = (float*)(ws + take((size_t)NN * FF1 * 4));   // 64MB (out1, then out2)
  unsigned short* h1bf = (unsigned short*)(ws + take((size_t)NN * FF1 * 2)); // 32MB
  float* deg    = (float*)(ws + take((size_t)NN * 4));
  float* loopea = (float*)(ws + take((size_t)NN * DEDGE * 4));
  unsigned short* xbf   = (unsigned short*)(ws + take((size_t)NN * DNODE * 2));
  unsigned short* wl1bf = (unsigned short*)(ws + take((size_t)FF1 * DNODE * 2));
  unsigned short* wr1bf = (unsigned short*)(ws + take((size_t)FF1 * DNODE * 2));
  unsigned short* wl2bf = (unsigned short*)(ws + take((size_t)FF2 * FF1 * 2));
  unsigned short* wr2bf = (unsigned short*)(ws + take((size_t)FF2 * FF1 * 2));
  float* logits = (float*)(ws + take((size_t)ETOT * NH1 * 4));
  unsigned int* mxkey = (unsigned int*)(ws + take((size_t)NN * NH1 * 4));
  float* den    = (float*)(ws + take((size_t)NN * NH1 * 4));
  float* pooled = (float*)(ws + take((size_t)GG * FF2 * 4));
  float* cnt    = (float*)(ws + take((size_t)GG * 4));
  float* m1     = (float*)(ws + take((size_t)GG * 512 * 4));
  float* m2     = (float*)(ws + take((size_t)GG * 256 * 4));
  float* xl2 = xl1;                                   // reuse after layer-1 scatter completes
  float* xr2 = (float*)((char*)xl1 + (size_t)NN * FF2 * 4);

  const int T = 256;

  // --- self-loop edge attrs (fill_value='mean') ---
  zero_f32<<<cdiv(NN, T), T, 0, stream>>>(deg, NN);
  zero_f32<<<cdiv((size_t)NN * DEDGE, T), T, 0, stream>>>(loopea, (size_t)NN * DEDGE);
  selfloop_accum<<<cdiv((size_t)EE * DEDGE, T), T, 0, stream>>>(dstI, ea, loopea, deg);
  selfloop_div<<<cdiv((size_t)NN * DEDGE, T), T, 0, stream>>>(loopea, deg);

  // --- bf16 conversions for WMMA GEMMs ---
  f32_to_bf16<<<cdiv((size_t)NN * DNODE, T), T, 0, stream>>>(x, xbf, (size_t)NN * DNODE);
  f32_to_bf16<<<cdiv((size_t)FF1 * DNODE, T), T, 0, stream>>>(Wl1, wl1bf, (size_t)FF1 * DNODE);
  f32_to_bf16<<<cdiv((size_t)FF1 * DNODE, T), T, 0, stream>>>(Wr1, wr1bf, (size_t)FF1 * DNODE);
  f32_to_bf16<<<cdiv((size_t)FF2 * FF1, T), T, 0, stream>>>(Wl2, wl2bf, (size_t)FF2 * FF1);
  f32_to_bf16<<<cdiv((size_t)FF2 * FF1, T), T, 0, stream>>>(Wr2, wr2bf, (size_t)FF2 * FF1);

  // --- layer 1: xl1/xr1 GEMMs (v_wmma_f32_16x16x32_bf16, 16x64 wave tiles) ---
  {
    unsigned ntiles = (NN / 16) * (FF1 / 64);
    gemm_bf16_wmma<<<cdiv(ntiles, 8), 256, 0, stream>>>(xbf, wl1bf, bl1, xl1, NN, FF1, DNODE, 0);
    gemm_bf16_wmma<<<cdiv(ntiles, 8), 256, 0, stream>>>(xbf, wr1bf, br1, xr1, NN, FF1, DNODE, 0);
  }
  zero_f32<<<cdiv((size_t)NN * FF1, T), T, 0, stream>>>(outb, (size_t)NN * FF1);
  zero_f32<<<cdiv((size_t)NN * NH1, T), T, 0, stream>>>((float*)mxkey, (size_t)NN * NH1);
  zero_f32<<<cdiv((size_t)NN * NH1, T), T, 0, stream>>>(den, (size_t)NN * NH1);

  gat_logits<<<cdiv((size_t)ETOT * NH1 * 32, T), T, 0, stream>>>(
      xl1, xr1, srcI, dstI, ea, loopea, We1, att1, logits, mxkey, NH1, FF1);
  gat_exps<<<cdiv((size_t)ETOT * NH1, T), T, 0, stream>>>(logits, mxkey, den, dstI, NH1);
  gat_accum<<<cdiv((size_t)ETOT * NH1 * 32, T), T, 0, stream>>>(
      xl1, logits, den, srcI, dstI, outb, NH1, FF1);
  bias_relu_bf16<<<cdiv((size_t)NN * FF1, T), T, 0, stream>>>(outb, bias1, h1bf,
                                                              (size_t)NN * FF1, FF1);

  // --- layer 2: xl2/xr2 GEMMs (K=2048, dominant WMMA work) ---
  {
    unsigned ntiles = (NN / 16) * (FF2 / 64);
    gemm_bf16_wmma<<<cdiv(ntiles, 8), 256, 0, stream>>>(h1bf, wl2bf, bl2, xl2, NN, FF2, FF1, 0);
    gemm_bf16_wmma<<<cdiv(ntiles, 8), 256, 0, stream>>>(h1bf, wr2bf, br2, xr2, NN, FF2, FF1, 0);
  }
  zero_f32<<<cdiv((size_t)NN * FF2, T), T, 0, stream>>>(outb, (size_t)NN * FF2);
  zero_f32<<<cdiv((size_t)NN * NH2, T), T, 0, stream>>>((float*)mxkey, (size_t)NN * NH2);
  zero_f32<<<cdiv((size_t)NN * NH2, T), T, 0, stream>>>(den, (size_t)NN * NH2);

  gat_logits<<<cdiv((size_t)ETOT * NH2 * 32, T), T, 0, stream>>>(
      xl2, xr2, srcI, dstI, ea, loopea, We2, att2, logits, mxkey, NH2, FF2);
  gat_exps<<<cdiv((size_t)ETOT * NH2, T), T, 0, stream>>>(logits, mxkey, den, dstI, NH2);
  gat_accum<<<cdiv((size_t)ETOT * NH2 * 32, T), T, 0, stream>>>(
      xl2, logits, den, srcI, dstI, outb, NH2, FF2);
  bias_add_inplace<<<cdiv((size_t)NN * FF2, T), T, 0, stream>>>(outb, bias2, (size_t)NN * FF2, FF2);

  // --- global mean pool + MLP head ---
  zero_f32<<<cdiv((size_t)GG * FF2, T), T, 0, stream>>>(pooled, (size_t)GG * FF2);
  zero_f32<<<1, T, 0, stream>>>(cnt, GG);
  pool_cnt<<<cdiv(NN, T), T, 0, stream>>>(batch, cnt);
  pool_sum<<<cdiv((size_t)NN * FF2, T), T, 0, stream>>>(outb, batch, pooled);
  pool_div<<<cdiv((size_t)GG * FF2, T), T, 0, stream>>>(pooled, cnt);

  linear_act<<<cdiv((size_t)GG * 512, T), T, 0, stream>>>(pooled, W1, b1, m1, GG, FF2, 512, 1);
  linear_act<<<cdiv((size_t)GG * 256, T), T, 0, stream>>>(m1, W2, b2, m2, GG, 512, 256, 1);
  linear_act<<<1, GG, 0, stream>>>(m2, W3, b3, (float*)d_out, GG, 256, 1, 2);
}
